// cal_EE_51427938402671
// MI455X (gfx1250) — matrix operand order repaired and checked by reading, not compiled, not run on hardware
//
#include <hip/hip_runtime.h>

typedef __attribute__((ext_vector_type(16))) _Float16 v16h;
typedef __attribute__((ext_vector_type(8)))  _Float16 v8h;
typedef __attribute__((ext_vector_type(8)))  float    v8f;

#define B_  4096
#define K_  32
#define L_  64
#define D_  256
#define N_  97      // L + K + 1
#define H1_ 128     // D/2

// ---------------------------------------------------------------------------
// Kernel 1: W1 [256][128] f32 -> w1t [128][256] f16 (transposed, K-contiguous)
// ---------------------------------------------------------------------------
__global__ void prep_w1(const float* __restrict__ W1, _Float16* __restrict__ w1t) {
    int idx = blockIdx.x * blockDim.x + threadIdx.x;   // 0..32767
    int n = idx >> 8;                                  // output col 0..127
    int k = idx & 255;                                 // k 0..255
    w1t[n * D_ + k] = (_Float16)W1[k * H1_ + n];
}

// ---------------------------------------------------------------------------
// Kernel 2: per-batch Laplacian-mean collapsed to weighted row sum.
// One block per batch, 256 threads (8 waves).
// Streaming phase: 4 groups x 64 lanes; lane owns 4 channels (b128 loads),
// group g takes rows m = g, g+4, ... (row index uniform per wave -> no div).
// ---------------------------------------------------------------------------
__global__ void __launch_bounds__(256) fuse_h(
        const float* __restrict__ uk, const float* __restrict__ uc,
        const float* __restrict__ rl, const float* __restrict__ e,
        const float* __restrict__ e_rc, _Float16* __restrict__ h16) {
    __shared__ float e_s[L_ * K_];      // 8 KB
    __shared__ float dinv_r[L_];
    __shared__ float dinv_u[K_];
    __shared__ float w_s[N_];
    __shared__ float misc[2];           // [0]=sum dinv_u, [1]=dinv_c
    __shared__ float part[4][D_];       // 4 KB cross-group reduce

    const int b = blockIdx.x;
    const int t = threadIdx.x;
    const float* eb = e + (size_t)b * (L_ * K_);

    // load e tile into LDS with b128
    {
        const float4* eb4 = (const float4*)eb;
        float4* es4 = (float4*)e_s;
        #pragma unroll
        for (int i = 0; i < (L_ * K_) / 4 / 256; ++i)
            es4[t + i * 256] = eb4[t + i * 256];
    }
    __syncthreads();

    if (t < L_) {                                   // row sums of e -> dinv for r-nodes
        float s = 0.f;
        for (int k = 0; k < K_; ++k) s += e_s[t * K_ + k];
        dinv_r[t] = rsqrtf(s);
    } else if (t < L_ + K_) {                       // col sums of e + K (ones) -> u-nodes
        int k = t - L_;
        float s = 0.f;
        for (int l = 0; l < L_; ++l) s += e_s[l * K_ + k];
        dinv_u[k] = rsqrtf(s + (float)K_);
    }
    __syncthreads();

    if (t == 0) {
        float su = 0.f;
        for (int k = 0; k < K_; ++k) su += dinv_u[k];
        misc[0] = su;
        const float* erc = e_rc + (size_t)b * K_;
        float sc = 0.f;
        for (int k = 0; k < K_; ++k) sc += erc[k];
        misc[1] = rsqrtf(sc + 1.f);                 // c-node rowsum = sum(e_rc) + 1
    }
    __syncthreads();

    const float invN = 1.f / (float)N_;
    if (t < L_) {                                   // w for r-rows
        float s = 0.f;
        for (int k = 0; k < K_; ++k) s += e_s[t * K_ + k] * dinv_u[k];
        w_s[t] = s * dinv_r[t] * invN;
    } else if (t < L_ + K_) {                       // w for u-rows
        int k = t - L_;
        float s = 0.f;
        for (int l = 0; l < L_; ++l) s += e_s[l * K_ + k] * dinv_r[l];
        w_s[L_ + k] = (s + misc[0]) * dinv_u[k] * invN;
    } else if (t == L_ + K_) {                      // w for c-row
        const float* erc = e_rc + (size_t)b * K_;
        float s = 0.f;
        for (int k = 0; k < K_; ++k) s += erc[k] * dinv_u[k];
        float dc = misc[1];
        w_s[N_ - 1] = (s + dc) * dc * invN;
    }
    __syncthreads();

    // ---- streaming weighted row sum (BW-bound part), b128 per lane ----
    const int tl = t & 63;          // channel-quarter lane
    const int g  = t >> 6;          // row group 0..3
    const float4* rl4 = (const float4*)(rl + (size_t)b * (L_ * D_));
    const float4* uk4 = (const float4*)(uk + (size_t)b * (K_ * D_));
    const float4* uc4 = (const float4*)(uc + (size_t)b * D_);

    float4 acc = make_float4(0.f, 0.f, 0.f, 0.f);
    for (int m = g; m < N_; m += 4) {
        float w = w_s[m];
        float4 x;
        if (m < L_)            x = rl4[m * (D_ / 4) + tl];
        else if (m < L_ + K_)  x = uk4[(m - L_) * (D_ / 4) + tl];
        else                   x = uc4[tl];
        acc.x += w * x.x; acc.y += w * x.y; acc.z += w * x.z; acc.w += w * x.w;
    }
    part[g][tl * 4 + 0] = acc.x;
    part[g][tl * 4 + 1] = acc.y;
    part[g][tl * 4 + 2] = acc.z;
    part[g][tl * 4 + 3] = acc.w;
    __syncthreads();

    float s = part[0][t] + part[1][t] + part[2][t] + part[3][t];
    h16[(size_t)b * D_ + t] = (_Float16)s;
}

// ---------------------------------------------------------------------------
// Kernel 3: MLP via WMMA. One wave per 16 batch rows.
//   h1 = relu(h @ W1 + b1) via v_wmma_f32_16x16x32_f16 (8 n-tiles x 8 k-steps)
//   logits = h1 @ W2 + b2 ; softmax over 2 classes.
// Fragment layouts per CDNA5 ISA:
//   A (16x32 f16): lane m(0..15) holds row m; elems 0..7 = K[0..7], 8..15 = K[16..23];
//                  lanes 16..31 hold K offsets +8 in each half.
//   B (32x16 f16): lane n(0..15) holds col n; elems 0..7 = K[0..7], 8..15 = K[16..23];
//                  lanes 16..31 hold K offsets +8 in each half.
// ---------------------------------------------------------------------------
__global__ void __launch_bounds__(32) mlp_wmma(
        const _Float16* __restrict__ h16, const _Float16* __restrict__ w1t,
        const float* __restrict__ b1, const float* __restrict__ W2,
        const float* __restrict__ b2, float* __restrict__ out) {
    __shared__ float h1s[16][H1_ + 4];
    __shared__ float lg[16][2];

    const int lane = threadIdx.x;        // 0..31
    const int m    = lane & 15;
    const int hi   = lane >> 4;          // 0 or 1 (lane half)
    const int row0 = blockIdx.x * 16;

    const _Float16* Abase = h16 + (size_t)(row0 + m) * D_ + hi * 8;

    for (int nt = 0; nt < 8; ++nt) {
        v8f acc = {};
        const _Float16* Bbase = w1t + (size_t)(nt * 16 + m) * D_ + hi * 8;
        #pragma unroll
        for (int ks = 0; ks < 8; ++ks) {
            const _Float16* Ap = Abase + ks * 32;
            v8h alo = *(const v8h*)(Ap);
            v8h ahi = *(const v8h*)(Ap + 16);
            v16h a;
            #pragma unroll
            for (int i = 0; i < 8; ++i) { a[i] = alo[i]; a[i + 8] = ahi[i]; }

            const _Float16* Bp = Bbase + ks * 32;
            v8h blo = *(const v8h*)(Bp);
            v8h bhi = *(const v8h*)(Bp + 16);
            v16h bb;
            #pragma unroll
            for (int i = 0; i < 8; ++i) { bb[i] = blo[i]; bb[i + 8] = bhi[i]; }

            acc = __builtin_amdgcn_wmma_f32_16x16x32_f16(
                false, a, false, bb, (short)0, acc, false, false);
        }
        // C/D layout: elem r -> (M = r + 8*hi, N = m)
        #pragma unroll
        for (int r = 0; r < 8; ++r) {
            int mrow = r + hi * 8;
            int col  = nt * 16 + m;
            float v  = acc[r] + b1[col];
            h1s[mrow][col] = v > 0.f ? v : 0.f;
        }
    }
    __syncthreads();

    // 32 lanes = 16 rows x 2 classes
    const int rr = lane >> 1;
    const int jj = lane & 1;
    float s = b2[jj];
    #pragma unroll 4
    for (int n = 0; n < H1_; ++n) s += h1s[rr][n] * W2[n * 2 + jj];
    lg[rr][jj] = s;
    __syncthreads();

    float l0 = lg[rr][0], l1 = lg[rr][1];
    float mx = fmaxf(l0, l1);
    float e0 = __expf(l0 - mx), e1 = __expf(l1 - mx);
    float mine = jj ? e1 : e0;
    out[(size_t)(row0 + rr) * 2 + jj] = mine / (e0 + e1);
}

// ---------------------------------------------------------------------------
extern "C" void kernel_launch(void* const* d_in, const int* in_sizes, int n_in,
                              void* d_out, int out_size, void* d_ws, size_t ws_size,
                              hipStream_t stream) {
    (void)in_sizes; (void)n_in; (void)out_size; (void)ws_size;
    const float* uk   = (const float*)d_in[0];
    const float* uc   = (const float*)d_in[1];
    const float* rl   = (const float*)d_in[2];
    const float* e    = (const float*)d_in[3];
    const float* e_rc = (const float*)d_in[4];
    const float* W1   = (const float*)d_in[5];
    const float* b1   = (const float*)d_in[6];
    const float* W2   = (const float*)d_in[7];
    const float* b2   = (const float*)d_in[8];
    float* out = (float*)d_out;

    _Float16* h16 = (_Float16*)d_ws;                                   // 2 MB
    _Float16* w1t = (_Float16*)((char*)d_ws + (size_t)B_ * D_ * 2);    // 64 KB

    prep_w1<<<(D_ * H1_) / 256, 256, 0, stream>>>(W1, w1t);
    fuse_h<<<B_, 256, 0, stream>>>(uk, uc, rl, e, e_rc, h16);
    mlp_wmma<<<B_ / 16, 32, 0, stream>>>(h16, w1t, b1, W2, b2, out);
}
